// ChaoticDecoder_1589137900161
// MI455X (gfx1250) — matrix operand order repaired and checked by reading, not compiled, not run on hardware
//
#include <hip/hip_runtime.h>
#include <hip/hip_bf16.h>

typedef __attribute__((ext_vector_type(16))) _Float16 v16h;
typedef __attribute__((ext_vector_type(8)))  _Float16 v8h;
typedef __attribute__((ext_vector_type(8)))  float    v8f;
typedef __attribute__((ext_vector_type(4)))  int      v4i;

// Address-space-qualified int4 for the async-to-LDS builtin signature.
typedef __attribute__((address_space(1))) v4i ga_v4i;   // global
typedef __attribute__((address_space(3))) v4i ls_v4i;   // LDS

#define BS   128
#define SEQ  64
#define DD   64
#define HH   128
#define OUTN 4
#define G4H  512

// Workspace offsets (bytes); all multiples of 32 for v16h/v8h alignment.
static constexpr size_t OFF_WAXT  = 0;                        // [64][64]   f16  (Wa[:D,:]^T)
static constexpr size_t OFF_WIT   = OFF_WAXT + 64*64*2;       // [512][64]  f16  (Wi^T)
static constexpr size_t OFF_WVIT  = OFF_WIT  + 512*64*2;      // [512][64]  f16  (Wvi^T)
static constexpr size_t OFF_WHT   = OFF_WVIT + 512*64*2;      // [512][128] f16  (Wh^T)
static constexpr size_t OFF_WVHT  = OFF_WHT  + 512*128*2;     // [512][128] f16  (Wvh^T)
static constexpr size_t OFF_CTX16 = OFF_WVHT + 512*128*2;     // [128][64]  f16  (context)

__device__ __forceinline__ v8f wmma16(v16h a, v16h b, v8f c) {
  // v_wmma_f32_16x16x32_f16: (neg_a, A, neg_b, B, c_mod, C, reuse_a, reuse_b)
  return __builtin_amdgcn_wmma_f32_16x16x32_f16(false, a, false, b, (short)0, c, false, false);
}
__device__ __forceinline__ float sigmoidf_fast(float x) { return 1.0f / (1.0f + __expf(-x)); }

// CDNA5 async memory->LDS copy (ASYNCcnt path), with portable fallback.
__device__ __forceinline__ void async_copy_b128(void* lds_dst, const void* gsrc) {
#if __has_builtin(__builtin_amdgcn_global_load_async_to_lds_b128)
  __builtin_amdgcn_global_load_async_to_lds_b128(
      (ga_v4i*)(v4i*)gsrc,      // generic -> addrspace(1) int4*
      (ls_v4i*)(v4i*)lds_dst,   // generic -> addrspace(3) int4*
      0, 0);
#else
  *(uint4*)lds_dst = *(const uint4*)gsrc;   // sync fallback: vmem load + ds store
#endif
}
__device__ __forceinline__ void wait_async_copies() {
#if __has_builtin(__builtin_amdgcn_s_wait_asynccnt)
  __builtin_amdgcn_s_wait_asynccnt(0);
#elif __has_builtin(__builtin_amdgcn_global_load_async_to_lds_b128)
  asm volatile("s_wait_asynccnt 0x0" ::: "memory");
#endif
}

// Assemble a 16-bit WMMA A fragment (16x32 MxK) from a row-major [16][Kstride] f16 array.
// Layout (matches B): low lanes hold K=[0..15], high lanes K=[16..31], contiguous.
__device__ __forceinline__ v16h load_a_frag(const _Float16* rowbase, int lane) {
  return *(const v16h*)(rowbase + ((lane >> 4) << 4));
}

// B fragment (32x16 KxN) from transposed weights stored [N][Kstride] f16:
// lane = column N, contiguous 16 K values (low lanes K=0..15, high lanes K=16..31).
__device__ __forceinline__ v16h load_b_frag(const _Float16* bt, int kstride, int ncol,
                                            int koff, int lane) {
  return *(const v16h*)(bt + (size_t)ncol * kstride + koff + (((lane >> 4) << 4)));
}

// ---------------------------------------------------------------------------
// Kernel 1: convert/transpose weights to f16 [N][K] layouts in workspace.
// ---------------------------------------------------------------------------
__global__ __launch_bounds__(256) void prep_kernel(const float* __restrict__ Wa,
    const float* __restrict__ Wi,  const float* __restrict__ Wh,
    const float* __restrict__ Wvi, const float* __restrict__ Wvh,
    char* __restrict__ ws) {
  _Float16* waxt = (_Float16*)(ws + OFF_WAXT);
  _Float16* wit  = (_Float16*)(ws + OFF_WIT);
  _Float16* wvit = (_Float16*)(ws + OFF_WVIT);
  _Float16* wht  = (_Float16*)(ws + OFF_WHT);
  _Float16* wvht = (_Float16*)(ws + OFF_WVHT);
  int tid = blockIdx.x * blockDim.x + threadIdx.x;
  int np  = gridDim.x * blockDim.x;
  // Wa is [D+4H][D] row-major; only the first D rows matter (softmax over seq
  // cancels the state-dependent logit term, which is constant along seq).
  for (int i = tid; i < 64 * 64; i += np) {
    int j = i >> 6, d = i & 63;
    waxt[i] = (_Float16)Wa[d * 64 + j];
  }
  for (int i = tid; i < 512 * 64; i += np) {
    int n = i >> 6, k = i & 63;
    wit[i]  = (_Float16)Wi[k * 512 + n];
    wvit[i] = (_Float16)Wvi[k * 512 + n];
  }
  for (int i = tid; i < 512 * 128; i += np) {
    int n = i >> 7, k = i & 127;
    wht[i]  = (_Float16)Wh[k * 512 + n];
    wvht[i] = (_Float16)Wvh[k * 512 + n];
  }
}

// ---------------------------------------------------------------------------
// Kernel 2: per-batch attention (step-invariant). One WG per batch element.
// logit[s,j] = x[b,s,:]·Wa_x[:,j] + ba[j]; softmax over s; context = Σ alpha*x.
// ---------------------------------------------------------------------------
__global__ __launch_bounds__(128) void attn_kernel(const float* __restrict__ x,
    const float* __restrict__ ba, char* __restrict__ ws) {
  __shared__ __align__(32) _Float16 xs[SEQ * DD];   // x[b] as f16, row-major [s][d]
  __shared__ float lg[SEQ * DD];                    // logits [s][j]
  const int b   = blockIdx.x;
  const int tid = threadIdx.x;
  const float* xb = x + (size_t)b * SEQ * DD;
  for (int i = tid; i < SEQ * DD; i += 128) xs[i] = (_Float16)xb[i];
  __syncthreads();

  const _Float16* waxt = (const _Float16*)(ws + OFF_WAXT);
  const int wave = tid >> 5, lane = tid & 31;
  const int mrow = lane & 15, mb8 = (lane >> 4) << 3;

  // A: 16 seq rows of x (this wave's M tile), K = d = 0..63 in two chunks.
  v16h a0 = load_a_frag(xs + (wave * 16 + mrow) * DD + 0,  lane);
  v16h a1 = load_a_frag(xs + (wave * 16 + mrow) * DD + 32, lane);
  #pragma unroll
  for (int j = 0; j < 4; ++j) {
    int ncol = j * 16 + (lane & 15);
    v16h b0 = load_b_frag(waxt, DD, ncol, 0,  lane);
    v16h b1 = load_b_frag(waxt, DD, ncol, 32, lane);
    float bias = ba[ncol];
    v8f c;
    #pragma unroll
    for (int r = 0; r < 8; ++r) c[r] = bias;
    c = wmma16(a0, b0, c);
    c = wmma16(a1, b1, c);
    #pragma unroll
    for (int r = 0; r < 8; ++r) lg[(wave * 16 + mb8 + r) * DD + ncol] = c[r];
  }
  __syncthreads();

  if (tid < DD) {                                   // one thread per output column j
    const int j = tid;
    float mx = -1e30f;
    for (int s = 0; s < SEQ; ++s) mx = fmaxf(mx, lg[s * DD + j]);
    float sum = 0.f, acc = 0.f;
    for (int s = 0; s < SEQ; ++s) {
      float e = __expf(lg[s * DD + j] - mx);
      sum += e;
      acc += e * xb[s * DD + j];
    }
    ((_Float16*)(ws + OFF_CTX16))[b * DD + j] = (_Float16)(acc / sum);
  }
}

// ---------------------------------------------------------------------------
// Kernel 3: 64-step bidirectional-cell recurrence + output head.
// 8 WGs x 16 batch rows, 8 wave32s (one WG per WGP).
//  - Wh^T (cell 0) async-DMA'd into LDS once; Wvh^T (cell 1) streamed from L2
//    so per-step B-fetch bandwidth is split across LDS and L2.
//  - Step-invariant gx tiles live in registers (wave w owns hidden chunk w).
// ---------------------------------------------------------------------------
__global__ __launch_bounds__(256) void rec_kernel(const float* __restrict__ bg,
    const float* __restrict__ bvg, const float* __restrict__ Wfc,
    const float* __restrict__ bfc, char* __restrict__ ws, float* __restrict__ out) {
  __shared__ __align__(32) _Float16 wlds[G4H][HH];  // Wh^T staged in LDS (128 KB)
  __shared__ __align__(32) _Float16 h16[2][16][HH]; // hidden state, f16 WMMA A (8 KB)
  __shared__ float cst[2][16][HH];                  // cell state, f32 (16 KB)

  const int tid  = threadIdx.x;
  const int wave = tid >> 5, lane = tid & 31;
  const int mrow = lane & 15, mb8 = (lane >> 4) << 3;
  const int b0   = blockIdx.x * 16;

  const _Float16* wit   = (const _Float16*)(ws + OFF_WIT);
  const _Float16* wvit  = (const _Float16*)(ws + OFF_WVIT);
  const _Float16* wvht  = (const _Float16*)(ws + OFF_WVHT);
  const _Float16* ctx16 = (const _Float16*)(ws + OFF_CTX16);

  // Kick off async Wh^T -> LDS staging first so the DMA overlaps Phase A WMMAs.
  {
    char*       dst = (char*)&wlds[0][0];
    const char* src = ws + OFF_WHT;
    for (int i = tid * 16; i < G4H * HH * 2; i += 256 * 16)
      async_copy_b128(dst + i, src + i);
  }
  __builtin_prefetch(wvht, 0, 1);                   // warm L2 path for cell 1

  {  // zero h and c
    float*    cflat = &cst[0][0][0];
    _Float16* hflat = &h16[0][0][0];
    for (int i = tid; i < 2 * 16 * HH; i += 256) { cflat[i] = 0.f; hflat[i] = (_Float16)0.f; }
  }

  // Phase A: gx = context @ Wi + b, kept in registers. Wave w computes exactly
  // the 2 cells x 4 gate tiles at columns gt*128 + w*16 that it consumes later.
  v16h a0 = load_a_frag(ctx16 + (size_t)(b0 + mrow) * DD + 0,  lane);
  v16h a1 = load_a_frag(ctx16 + (size_t)(b0 + mrow) * DD + 32, lane);
  v8f gx[2][4];
  #pragma unroll
  for (int cell = 0; cell < 2; ++cell) {
    const _Float16* wt   = cell ? wvit : wit;
    const float*    bias = cell ? bvg  : bg;
    #pragma unroll
    for (int gt = 0; gt < 4; ++gt) {
      int ncol = gt * HH + wave * 16 + (lane & 15);
      v16h bb0 = load_b_frag(wt, DD, ncol, 0,  lane);
      v16h bb1 = load_b_frag(wt, DD, ncol, 32, lane);
      float bs = bias[ncol];
      v8f c;
      #pragma unroll
      for (int r = 0; r < 8; ++r) c[r] = bs;
      c = wmma16(a0, bb0, c);
      c = wmma16(a1, bb1, c);
      gx[cell][gt] = c;
    }
  }
  wait_async_copies();   // Wh^T resident in LDS
  __syncthreads();

  // Phase B: 64 steps. Wave w owns hidden chunk kc=w (16 units) for BOTH cells,
  // so all four gates of a hidden unit land in the same lane/fragment slot.
  const int kc = wave;
  for (int t = 0; t < 64; ++t) {
    v8f acc[2][4];
    #pragma unroll
    for (int cell = 0; cell < 2; ++cell) {
      const _Float16* wt = (cell == 0) ? (const _Float16*)&wlds[0][0] : wvht;
      v16h A[4];
      #pragma unroll
      for (int q = 0; q < 4; ++q)
        A[q] = load_a_frag(&h16[cell][mrow][q * 32], lane);
      #pragma unroll
      for (int gt = 0; gt < 4; ++gt) {               // i | f | g | o
        int ncol = gt * HH + kc * 16 + (lane & 15);
        v8f c = gx[cell][gt];
        #pragma unroll
        for (int q = 0; q < 4; ++q)                  // K = 128 in four 32-chunks
          c = wmma16(A[q], load_b_frag(wt, HH, ncol, q * 32, lane), c);
        acc[cell][gt] = c;
      }
    }
    __syncthreads();  // all reads of h16 (state t-1) done before anyone writes state t
    #pragma unroll
    for (int cell = 0; cell < 2; ++cell) {
      #pragma unroll
      for (int r = 0; r < 8; ++r) {
        int m = mb8 + r;
        int k = kc * 16 + (lane & 15);
        float ig = sigmoidf_fast(acc[cell][0][r]);
        float fg = sigmoidf_fast(acc[cell][1][r]);
        float gg = tanhf(acc[cell][2][r]);
        float og = sigmoidf_fast(acc[cell][3][r]);
        float cn = fg * cst[cell][m][k] + ig * gg;
        cst[cell][m][k] = cn;
        h16[cell][m][k] = (_Float16)(og * tanhf(cn));
      }
    }
    __syncthreads();  // state t visible before next step's fragment loads
  }

  // Phase C: out = concat(h, hv) @ Wfc + bfc   ([16,256]@[256,4] — VALU dots)
  if (tid < 64) {
    int row = tid >> 2, o = tid & 3;
    float s = bfc[o];
    for (int k = 0; k < HH; ++k) s += (float)h16[0][row][k] * Wfc[k * OUTN + o];
    for (int k = 0; k < HH; ++k) s += (float)h16[1][row][k] * Wfc[(HH + k) * OUTN + o];
    out[(b0 + row) * OUTN + o] = s;
  }
}

// ---------------------------------------------------------------------------
extern "C" void kernel_launch(void* const* d_in, const int* in_sizes, int n_in,
                              void* d_out, int out_size, void* d_ws, size_t ws_size,
                              hipStream_t stream) {
  const float* x   = (const float*)d_in[0];
  const float* Wa  = (const float*)d_in[1];
  const float* ba  = (const float*)d_in[2];
  const float* Wi  = (const float*)d_in[3];
  const float* Wh  = (const float*)d_in[4];
  const float* b   = (const float*)d_in[5];
  const float* Wvi = (const float*)d_in[6];
  const float* Wvh = (const float*)d_in[7];
  const float* bv  = (const float*)d_in[8];
  const float* Wfc = (const float*)d_in[9];
  const float* bfc = (const float*)d_in[10];
  char*  ws  = (char*)d_ws;
  float* out = (float*)d_out;

  prep_kernel<<<208, 256, 0, stream>>>(Wa, Wi, Wh, Wvi, Wvh, ws);
  attn_kernel<<<BS, 128, 0, stream>>>(x, ba, ws);
  rec_kernel<<<BS / 16, 256, 0, stream>>>(b, bv, Wfc, bfc, ws, out);
}
